// TransformerMultiHeadAttention_6004364280104
// MI455X (gfx1250) — matrix operand order repaired and checked by reading, not compiled, not run on hardware
//
#include <hip/hip_runtime.h>

// ---------------------------------------------------------------------------
// MI455X (gfx1250, wave32) multi-head attention:
//   qp = q@Wq^T, kp = k@Wk^T, vp = v@Wv^T   (bf16 WMMA GEMM, fp32 accum)
//   flash-attention per (b,h) with causal + padding mask, async-to-LDS staging
//   out = attn_out @ Wo^T                   (bf16 WMMA GEMM, fp32 out)
// B=4, S=1024, D=1024, H=16, HD=64
// ---------------------------------------------------------------------------

typedef __attribute__((ext_vector_type(16))) __bf16 v16bf;
typedef __attribute__((ext_vector_type(8)))  float  v8f;

#define WMMA_BF16(a, b, c) \
  __builtin_amdgcn_wmma_f32_16x16x32_bf16(false, (a), false, (b), (short)0, (c), false, false)

static constexpr int S_LEN = 1024;
static constexpr int DMODEL = 1024;
static constexpr int NHEAD = 16;
static constexpr int HD = 64;
static constexpr float NEGINF = -1e9f;
static constexpr float SCALE = 0.125f; // 1/sqrt(64)

// ---- fragment loaders -------------------------------------------------------
// A fragment, 16x32 bf16 (M x K). Lane l<16: M=mbase+l, K in {0..7,16..23};
// lane l>=16: M=mbase+(l-16), K in {8..15,24..31}. VGPR j holds a K-pair.
__device__ __forceinline__ v16bf load_a_frag(const __bf16* base, int mbase,
                                             int ldk, int koff, int lane) {
  union { v16bf v; unsigned u[8]; } r;
  int m = mbase + (lane & 15);
  int khalf = (lane & 16) ? 8 : 0;
  const unsigned* rowp = (const unsigned*)(base + (size_t)m * ldk + koff);
#pragma unroll
  for (int j = 0; j < 8; ++j) {
    int kb = ((j < 4) ? 2 * j : 16 + 2 * (j - 4)) + khalf;
    r.u[j] = rowp[kb >> 1];
  }
  return r.v;
}

// B fragment, 32x16 bf16 (K x N), stored as base[n*ldk + k].
// Lane l<16: N=nbase+l, K in {0..7,16..23}; lane l>=16: N=nbase+(l-16), K in {8..15,24..31}.
__device__ __forceinline__ v16bf load_b_frag(const __bf16* base, int nbase,
                                             int ldk, int koff, int lane) {
  union { v16bf v; unsigned u[8]; } r;
  int n = nbase + (lane & 15);
  int khalf = (lane & 16) ? 8 : 0;
  const unsigned* rowp = (const unsigned*)(base + (size_t)n * ldk + koff);
#pragma unroll
  for (int j = 0; j < 8; ++j) r.u[j] = rowp[(((j < 4) ? 2 * j : 16 + 2 * (j - 4)) + khalf) >> 1];
  return r.v;
}

__device__ __forceinline__ unsigned pack_bf16x2(float x, float y) {
  union { __bf16 h[2]; unsigned u; } c;
  c.h[0] = (__bf16)x; c.h[1] = (__bf16)y;
  return c.u;
}

__device__ __forceinline__ uint4 pack_bf16x8(const float4& a, const float4& b) {
  uint4 r;
  r.x = pack_bf16x2(a.x, a.y); r.y = pack_bf16x2(a.z, a.w);
  r.z = pack_bf16x2(b.x, b.y); r.w = pack_bf16x2(b.z, b.w);
  return r;
}

// CDNA5 async copy: global -> LDS (16B per lane), tracked by ASYNCcnt.
__device__ __forceinline__ void async_copy_b128(const __bf16* gsrc,
                                                void* lds_dst) {
  unsigned ldsAddr = (unsigned)(size_t)lds_dst;  // addr[31:0] = LDS offset
  asm volatile("global_load_async_to_lds_b128 %0, %1, off"
               :: "v"(ldsAddr), "v"(gsrc)
               : "memory");
}
__device__ __forceinline__ void async_wait0() {
  asm volatile("s_wait_asynccnt 0x0" ::: "memory");
}

// ---------------------------------------------------------------------------
// GEMM: C[M,N] = X[M,K] @ W[N,K]^T.  BM=128 BN=256 BK=32, 256 thr (8 waves),
// each wave computes a 64x64 tile (16 WMMAs per K-step).
// mode 0: bf16 out, [B,H,S,HD]  (Q, K projections)
// mode 2: bf16 out, [B,H,HD,S]  (V projection, pre-transposed for attention)
// mode 1: fp32 out, row-major [M,N]  (final projection)
// ---------------------------------------------------------------------------
__global__ __launch_bounds__(256)
void mha_gemm_bf16(const float* __restrict__ X, const float* __restrict__ W,
                   void* __restrict__ out, int M, int N, int K, int mode) {
  __shared__ alignas(16) __bf16 As[128 * 32];
  __shared__ alignas(16) __bf16 Bs[256 * 32];

  const int tid = threadIdx.x;
  const int lane = tid & 31;
  const int wave = tid >> 5;
  const int wm = wave & 1;   // 0..1 : 64-row half
  const int wn = wave >> 1;  // 0..3 : 64-col slice
  const int m0 = blockIdx.x * 128;
  const int n0 = blockIdx.y * 256;

  const int arow = tid >> 1;           // 0..127
  const int acg = (tid & 1) * 16;      // 0/16

  v8f acc[4][4] = {};

  for (int k0 = 0; k0 < K; k0 += 32) {
    // stage X tile -> As (fp32 -> bf16): 16 floats / thread
    {
      const float* src = X + (size_t)(m0 + arow) * K + k0 + acg;
      float4 f0 = ((const float4*)src)[0];
      float4 f1 = ((const float4*)src)[1];
      float4 f2 = ((const float4*)src)[2];
      float4 f3 = ((const float4*)src)[3];
      uint4* dst = (uint4*)(As + arow * 32 + acg);
      dst[0] = pack_bf16x8(f0, f1);
      dst[1] = pack_bf16x8(f2, f3);
    }
    // stage W tile -> Bs: one 32-float row of W per thread
    {
      const float* src = W + (size_t)(n0 + tid) * K + k0;
      uint4* dst = (uint4*)(Bs + tid * 32);
#pragma unroll
      for (int g = 0; g < 4; ++g) {
        float4 fa = ((const float4*)src)[2 * g];
        float4 fb = ((const float4*)src)[2 * g + 1];
        dst[g] = pack_bf16x8(fa, fb);
      }
    }
    if (k0 + 32 < K) {
      __builtin_prefetch(X + (size_t)(m0 + arow) * K + k0 + 32 + acg, 0, 1);
      __builtin_prefetch(W + (size_t)(n0 + tid) * K + k0 + 32, 0, 1);
    }
    __syncthreads();

    v16bf af[4], bf_[4];
#pragma unroll
    for (int mt = 0; mt < 4; ++mt)
      af[mt] = load_a_frag(As, wm * 64 + mt * 16, 32, 0, lane);
#pragma unroll
    for (int nt = 0; nt < 4; ++nt)
      bf_[nt] = load_b_frag(Bs, wn * 64 + nt * 16, 32, 0, lane);
#pragma unroll
    for (int mt = 0; mt < 4; ++mt)
#pragma unroll
      for (int nt = 0; nt < 4; ++nt)
        acc[mt][nt] = WMMA_BF16(af[mt], bf_[nt], acc[mt][nt]);
    __syncthreads();
  }

  // epilogue
  const int rowadd = (lane & 16) ? 8 : 0;
  const int coll = lane & 15;
#pragma unroll
  for (int mt = 0; mt < 4; ++mt) {
#pragma unroll
    for (int nt = 0; nt < 4; ++nt) {
#pragma unroll
      for (int i = 0; i < 8; ++i) {
        int row = m0 + wm * 64 + mt * 16 + i + rowadd;
        int col = n0 + wn * 64 + nt * 16 + coll;
        float val = acc[mt][nt][i];
        int b = row >> 10, s = row & 1023, h = col >> 6, hd = col & 63;
        if (mode == 0) {        // [B,H,S,HD] bf16
          ((__bf16*)out)[((((size_t)b * NHEAD + h) * S_LEN) + s) * HD + hd] =
              (__bf16)val;
        } else if (mode == 2) { // [B,H,HD,S] bf16 (transposed V)
          ((__bf16*)out)[((((size_t)b * NHEAD + h) * HD) + hd) * S_LEN + s] =
              (__bf16)val;
        } else {                // fp32 row-major
          ((float*)out)[(size_t)row * N + col] = val;
        }
      }
    }
  }
}

// ---------------------------------------------------------------------------
// Flash attention. Block = 256 thr (8 waves), handles one (b,h) x 128 q-rows.
// Each wave owns 16 query rows. KV in tiles of 32 keys, staged via
// global_load_async_to_lds_b128 (ASYNCcnt).
// Qp/Kp: [B,H,S,HD] bf16;  Vp: [B,H,HD,S] bf16 (pre-transposed).
// ---------------------------------------------------------------------------
__global__ __launch_bounds__(256)
void mha_flash_attn(const __bf16* __restrict__ Qp, const __bf16* __restrict__ Kp,
                    const __bf16* __restrict__ Vp,
                    const unsigned char* __restrict__ pad,
                    float* __restrict__ attn_out) {
  __shared__ alignas(16) __bf16 Ks[32 * 64];     // [key][hd]
  __shared__ alignas(16) __bf16 Vt[64 * 32];     // [hd][key]
  __shared__ alignas(16) __bf16 Ps[8][16 * 32];  // per-wave P scratch

  const int tid = threadIdx.x;
  const int lane = tid & 31;
  const int wave = tid >> 5;
  const int qblock = blockIdx.x * 128;
  const int bh = blockIdx.y;  // b*16 + h
  const int b = bh >> 4;
  const int h = bh & 15;
  const int q0 = qblock + wave * 16;

  const __bf16* Qb = Qp + (size_t)bh * S_LEN * HD;
  const __bf16* Kb = Kp + (size_t)bh * S_LEN * HD;
  const __bf16* Vb = Vp + (size_t)bh * HD * S_LEN;  // [hd][s]

  // Q fragments for this wave's 16 rows (hd chunks 0..31, 32..63)
  v16bf aq[2];
  aq[0] = load_a_frag(Qb, q0, HD, 0, lane);
  aq[1] = load_a_frag(Qb, q0, HD, 32, lane);

  v8f o[4] = {};
  float mrow[8], lrow[8];
#pragma unroll
  for (int i = 0; i < 8; ++i) { mrow[i] = -1e30f; lrow[i] = 0.0f; }

  const int rowadd = (lane & 16) ? 8 : 0;
  const int coll = lane & 15;

  // staging coordinates for the V tile: 64 hd-rows x 32 keys, 16B per thread
  const int vhd = tid >> 2;
  const int vgrp = (tid & 3) * 8;

  for (int n0 = 0; n0 < S_LEN; n0 += 32) {
    if (n0 > qblock + 127) break;  // causal early-out (block-uniform)

    // async stage K tile (32 keys x 64 hd = 4KB): tid*8 elements each
    async_copy_b128(Kb + (size_t)n0 * HD + tid * 8, Ks + tid * 8);
    // async stage V tile from transposed layout: row hd, keys n0..n0+31
    async_copy_b128(Vb + (size_t)vhd * S_LEN + n0 + vgrp,
                    Vt + vhd * 32 + vgrp);
    async_wait0();
    __syncthreads();

    // scores: two 16x16 tiles (keys n0..+15, n0+16..+31)
    v8f s0 = {}, s1 = {};
#pragma unroll
    for (int c = 0; c < 2; ++c) {
      v16bf bk0 = load_b_frag(Ks, 0, HD, c * 32, lane);
      v16bf bk1 = load_b_frag(Ks, 16, HD, c * 32, lane);
      s0 = WMMA_BF16(aq[c], bk0, s0);
      s1 = WMMA_BF16(aq[c], bk1, s1);
    }

    const int key0 = n0 + coll;
    const int key1 = n0 + 16 + coll;
    const bool pad0 = pad[(size_t)b * S_LEN + key0] != 0;
    const bool pad1 = pad[(size_t)b * S_LEN + key1] != 0;

#pragma unroll
    for (int i = 0; i < 8; ++i) {
      const int q = q0 + i + rowadd;
      float v0 = s0[i] * SCALE;
      float v1 = s1[i] * SCALE;
      if (pad0 || key0 > q) v0 = NEGINF;
      if (pad1 || key1 > q) v1 = NEGINF;

      // row max across the 16-lane N-half (wave32 xor shuffles stay in-half)
      float t = fmaxf(v0, v1);
      t = fmaxf(t, __shfl_xor(t, 1));
      t = fmaxf(t, __shfl_xor(t, 2));
      t = fmaxf(t, __shfl_xor(t, 4));
      t = fmaxf(t, __shfl_xor(t, 8));
      float mnew = fmaxf(mrow[i], t);

      float p0 = __expf(v0 - mnew);
      float p1 = __expf(v1 - mnew);
      float rs = p0 + p1;
      rs += __shfl_xor(rs, 1);
      rs += __shfl_xor(rs, 2);
      rs += __shfl_xor(rs, 4);
      rs += __shfl_xor(rs, 8);

      float alpha = __expf(mrow[i] - mnew);
      lrow[i] = lrow[i] * alpha + rs;
      mrow[i] = mnew;
#pragma unroll
      for (int t4 = 0; t4 < 4; ++t4) o[t4][i] *= alpha;

      // stash P (bf16) row-major into this wave's LDS scratch
      __bf16* prow = &Ps[wave][(i + rowadd) * 32];
      prow[coll] = (__bf16)p0;
      prow[16 + coll] = (__bf16)p1;
    }

    // lanes run in lockstep within the wave; just drain the DS pipe
    asm volatile("s_wait_dscnt 0" ::: "memory");

    // P (16x32) as A fragment, V^T slices as B fragments -> O += P.V
    v16bf ap = load_a_frag(&Ps[wave][0], 0, 32, 0, lane);
#pragma unroll
    for (int t4 = 0; t4 < 4; ++t4) {
      v16bf bv = load_b_frag(Vt, t4 * 16, 32, 0, lane);
      o[t4] = WMMA_BF16(ap, bv, o[t4]);
    }
    __syncthreads();
  }

  // normalize + write fp32 attn_out in [B,S,D] layout
#pragma unroll
  for (int i = 0; i < 8; ++i) {
    float inv = 1.0f / lrow[i];
    int q = q0 + i + rowadd;
#pragma unroll
    for (int t4 = 0; t4 < 4; ++t4) {
      int d = h * HD + t4 * 16 + coll;
      attn_out[((size_t)b * S_LEN + q) * DMODEL + d] = o[t4][i] * inv;
    }
  }
}

// ---------------------------------------------------------------------------
extern "C" void kernel_launch(void* const* d_in, const int* in_sizes, int n_in,
                              void* d_out, int out_size, void* d_ws,
                              size_t ws_size, hipStream_t stream) {
  (void)in_sizes; (void)n_in; (void)out_size; (void)ws_size;
  const float* q = (const float*)d_in[0];
  const float* k = (const float*)d_in[1];
  const float* v = (const float*)d_in[2];
  const unsigned char* pad = (const unsigned char*)d_in[3];
  // d_in[4] = causal_mask (reproduced analytically), d_in[9] = n_heads
  const float* Wq = (const float*)d_in[5];
  const float* Wk = (const float*)d_in[6];
  const float* Wv = (const float*)d_in[7];
  const float* Wo = (const float*)d_in[8];

  const int M = 4 * S_LEN;  // 4096 tokens
  char* ws = (char*)d_ws;
  const size_t projBytes = (size_t)M * DMODEL * sizeof(__bf16);  // 8 MB
  __bf16* Qp = (__bf16*)(ws);
  __bf16* Kp = (__bf16*)(ws + projBytes);
  __bf16* Vp = (__bf16*)(ws + 2 * projBytes);  // transposed [B,H,HD,S]
  float* attn = (float*)(ws + 3 * projBytes);  // 16 MB fp32

  dim3 gg(M / 128, DMODEL / 256);  // 32 x 4
  mha_gemm_bf16<<<gg, 256, 0, stream>>>(q, Wq, Qp, M, DMODEL, DMODEL, 0);
  mha_gemm_bf16<<<gg, 256, 0, stream>>>(k, Wk, Kp, M, DMODEL, DMODEL, 0);
  mha_gemm_bf16<<<gg, 256, 0, stream>>>(v, Wv, Vp, M, DMODEL, DMODEL, 2);

  dim3 ga(S_LEN / 128, 4 * NHEAD);  // 8 x 64
  mha_flash_attn<<<ga, 256, 0, stream>>>(Qp, Kp, Vp, pad, attn);

  mha_gemm_bf16<<<gg, 256, 0, stream>>>(attn, Wo, d_out, M, DMODEL, DMODEL, 1);
}
